// STDGN_45569603010736
// MI455X (gfx1250) — matrix operand order repaired and checked by reading, not compiled, not run on hardware
//
#include <hip/hip_runtime.h>
#include <hip/hip_bf16.h>

// ---- problem constants (match reference) ----
#define BB   16
#define SS   16
#define CC   8
#define HH   40
#define WW   40
#define NN   1600          // H*W
#define HID  128
#define OUTD 4
#define KNN  8
#define G3   384           // 3*HID
#define NT   100           // NN/16 row tiles per (b,s)
#define NWORDS 50          // 1600/32
#define MAXDEG 40

typedef __attribute__((ext_vector_type(16))) _Float16 v16h;
typedef __attribute__((ext_vector_type(8)))  float    v8f;

// ---------------- WMMA fragment helpers (wave32, gfx1250 layouts) ----------------
// A (16x32 f16): lane m=lane&15, half=lane>>4; VGPR v<4: K=half*8+2v ; v>=4: K=16+half*8+2(v-4)
__device__ __forceinline__ v16h load_a_f16(const _Float16* base, int lda, int k0) {
    int lane = threadIdx.x & 31;
    int m = lane & 15, half = lane >> 4;
    const _Float16* p = base + (long)m * lda + k0 + half * 8;
    v16h a;
#pragma unroll
    for (int v = 0; v < 8; ++v) {
        int o = (v < 4) ? (2 * v) : (16 + 2 * (v - 4));
        a[2 * v]     = p[o];
        a[2 * v + 1] = p[o + 1];
    }
    return a;
}

// B (32x16 f16) from K-contiguous (transposed) storage bt[j*ldb + k]:
// lane: N=j0+(lane&15), half=lane>>4; VGPR v<4: K=half*8+2v ; v>=4: K=16+half*8+2(v-4)
__device__ __forceinline__ v16h load_b_t(const _Float16* bt, int ldb, int j0, int k0) {
    int lane = threadIdx.x & 31;
    int j = j0 + (lane & 15);
    int kb = k0 + (lane >> 4) * 8;
    const _Float16* p = bt + (long)j * ldb + kb;
    v16h b;
#pragma unroll
    for (int v = 0; v < 8; ++v) {
        int o = (v < 4) ? (2 * v) : (16 + 2 * (v - 4));
        b[2 * v]     = p[o];
        b[2 * v + 1] = p[o + 1];
    }
    return b;
}

__device__ __forceinline__ v8f wmma16(v16h a, v16h b, v8f c) {
    return __builtin_amdgcn_wmma_f32_16x16x32_f16(false, a, false, b, (short)0, c, false, false);
}

// ---------------- adjacency construction (exact (dist,index)-stable kNN) ----------------
__global__ void k_adj_clear(unsigned* bits) {
    int i = blockIdx.x * blockDim.x + threadIdx.x;
    if (i < NN * NWORDS) bits[i] = 0u;
}

__global__ void k_adj_knn(unsigned* bits) {
    int n = blockIdx.x * blockDim.x + threadIdx.x;
    if (n >= NN) return;
    int r = n / WW, c = n % WW;
    int cnt = 0;
    for (int d = 1; d <= 8 && cnt < KNN; ++d) {
        for (int m = 0; m < NN && cnt < KNN; ++m) {
            int mr = m / WW, mc = m % WW;
            int dist = abs(mr - r) + abs(mc - c);
            if (dist == d) {
                atomicOr(&bits[(long)n * NWORDS + (m >> 5)], 1u << (m & 31));
                atomicOr(&bits[(long)m * NWORDS + (n >> 5)], 1u << (n & 31));
                ++cnt;
            }
        }
    }
}

__global__ void k_adj_csr(const unsigned* bits, int* nbr, int* deg) {
    int n = blockIdx.x * blockDim.x + threadIdx.x;
    if (n >= NN) return;
    int d = 0;
    nbr[n * MAXDEG + d++] = n;  // self loop (+I)
    for (int w = 0; w < NWORDS; ++w) {
        unsigned v = bits[(long)n * NWORDS + w];
        while (v) {
            int b = __ffs(v) - 1;
            v &= v - 1;
            int m = w * 32 + b;
            if (m != n && d < MAXDEG) nbr[n * MAXDEG + d++] = m;
        }
    }
    deg[n] = d;
}

// ---------------- weight prep ----------------
__global__ void k_cvt_f16(const float* src, _Float16* dst, int n) {
    int i = blockIdx.x * blockDim.x + threadIdx.x;
    if (i < n) dst[i] = (_Float16)src[i];
}

__global__ void k_transpose_w2(const float* w2, _Float16* w2t) {  // w2t[j][k] = w2[k][j]
    int i = blockIdx.x * blockDim.x + threadIdx.x;
    if (i < HID * HID) {
        int j = i >> 7, k = i & 127;
        w2t[i] = (_Float16)w2[k * HID + j];
    }
}

__global__ void k_zero_f32(float* p, int n) {
    int i = blockIdx.x * blockDim.x + threadIdx.x;
    if (i < n) p[i] = 0.f;
}

__global__ void k_zero_f16(_Float16* p, int n) {
    int i = blockIdx.x * blockDim.x + threadIdx.x;
    if (i < n) p[i] = (_Float16)0.f;
}

// ---------------- GCN layer 1: sparse gather + 8->128 matvec ----------------
__global__ void k_gcn1(const float* __restrict__ x, const float* __restrict__ w1,
                       const float* __restrict__ b1, const int* __restrict__ nbr,
                       const int* __restrict__ deg, _Float16* __restrict__ h1h) {
    int blk = blockIdx.x;              // over B*S*N
    int bs = blk / NN, n = blk % NN;
    __shared__ float aggx[CC];
    int tid = threadIdx.x;
    if (tid < CC) {
        float acc = 0.f;
        int d = deg[n];
        const float* xp = x + ((long)bs * CC + tid) * NN;
        for (int i = 0; i < d; ++i) acc += xp[nbr[n * MAXDEG + i]];
        aggx[tid] = acc;
    }
    __syncthreads();
    float acc = b1[tid];
#pragma unroll
    for (int c = 0; c < CC; ++c) acc += aggx[c] * w1[c * HID + tid];
    h1h[(long)blk * HID + tid] = (_Float16)(acc > 0.f ? acc : 0.f);
}

// ---------------- sparse aggregation of h1 (128 channels) ----------------
__global__ void k_gcn_agg(const _Float16* __restrict__ h1h, const int* __restrict__ nbr,
                          const int* __restrict__ deg, _Float16* __restrict__ agg2h) {
    int blk = blockIdx.x;              // over B*S*N
    int bs = blk / NN, n = blk % NN;
    int j = threadIdx.x;
    float acc = 0.f;
    int d = deg[n];
    const _Float16* hp = h1h + (long)bs * NN * HID;
    for (int i = 0; i < d; ++i) acc += (float)hp[(long)nbr[n * MAXDEG + i] * HID + j];
    agg2h[(long)blk * HID + j] = (_Float16)acc;
}

// ---------------- layer2 GEMM: feats = agg2 @ W2 + b2 (WMMA), [B,S,N]->[S,B,N] ----------------
__global__ void k_gemm_feats(const _Float16* __restrict__ agg2h, const _Float16* __restrict__ w2t,
                             const float* __restrict__ b2, _Float16* __restrict__ featsh) {
    int rowtile = blockIdx.x;          // over B*S*NT
    int wv = threadIdx.x >> 5;         // 8 waves = 8 col tiles
    int b = rowtile / (SS * NT);
    int rem = rowtile % (SS * NT);
    int s = rem / NT, nt = rem % NT;
    long irow = ((long)(b * SS + s) * NN + nt * 16);
    const _Float16* Abase = agg2h + irow * HID;
    v8f acc = {};
#pragma unroll
    for (int kt = 0; kt < 4; ++kt) {
        v16h a = load_a_f16(Abase, HID, kt * 32);
        v16h bb = load_b_t(w2t, HID, wv * 16, kt * 32);
        acc = wmma16(a, bb, acc);
    }
    long orow = ((long)(s * BB + b) * NN + nt * 16);
    int lane = threadIdx.x & 31;
    int half = lane >> 4, col = wv * 16 + (lane & 15);
    float bias = b2[col];
#pragma unroll
    for (int r = 0; r < 8; ++r) {
        int mr = r + half * 8;
        featsh[(orow + mr) * HID + col] = (_Float16)(acc[r] + bias);
    }
}

// ---------------- GRU step: gi/gh WMMA + fused gates ----------------
// hin/hout: fp32 recurrent state. hfin/hfout: f16 shadow used as the WMMA A operand
// (written alongside hout so the next step's gh GEMM uses vectorized f16 loads).
__global__ void k_gru_step(const _Float16* __restrict__ featsh, const float* __restrict__ hin,
                           const _Float16* __restrict__ hfin, float* __restrict__ hout,
                           _Float16* __restrict__ hfout, const _Float16* __restrict__ wihh,
                           const _Float16* __restrict__ whhh, const float* __restrict__ b_ih,
                           const float* __restrict__ b_hh, int t) {
    __shared__ float tg[6][16][16];    // gates: 0..2 = gi(r,z,n), 3..5 = gh(r,z,n)
    int blk = blockIdx.x;              // rowtiles(B*N/16) * jtiles(8)
    int rowtile = blk >> 3, jt = blk & 7;
    long rowbase = (long)rowtile * 16; // in B*N row space
    int wv = threadIdx.x >> 5;         // 2 waves
    int lane = threadIdx.x & 31;
    int half = lane >> 4, c = lane & 15;
    if (wv == 0) {
        const _Float16* Abase = featsh + ((long)t * (BB * NN) + rowbase) * HID;
#pragma unroll
        for (int g = 0; g < 3; ++g) {
            v8f acc = {};
            int col0 = g * HID + jt * 16;
#pragma unroll
            for (int kt = 0; kt < 4; ++kt) {
                v16h a = load_a_f16(Abase, HID, kt * 32);
                v16h bb = load_b_t(wihh, HID, col0, kt * 32);
                acc = wmma16(a, bb, acc);
            }
#pragma unroll
            for (int r = 0; r < 8; ++r) tg[g][r + half * 8][c] = acc[r];
        }
    } else {
        const _Float16* Hbase = hfin + rowbase * HID;
#pragma unroll
        for (int g = 0; g < 3; ++g) {
            v8f acc = {};
            int col0 = g * HID + jt * 16;
#pragma unroll
            for (int kt = 0; kt < 4; ++kt) {
                v16h a = load_a_f16(Hbase, HID, kt * 32);
                v16h bb = load_b_t(whhh, HID, col0, kt * 32);
                acc = wmma16(a, bb, acc);
            }
#pragma unroll
            for (int r = 0; r < 8; ++r) tg[3 + g][r + half * 8][c] = acc[r];
        }
    }
    __syncthreads();
    for (int e = threadIdx.x; e < 256; e += 64) {
        int mr = e >> 4, cc = e & 15;
        int j = jt * 16 + cc;
        long hi = (rowbase + mr) * HID + j;
        float ir = tg[0][mr][cc] + b_ih[j];
        float iz = tg[1][mr][cc] + b_ih[HID + j];
        float in_ = tg[2][mr][cc] + b_ih[2 * HID + j];
        float hr = tg[3][mr][cc] + b_hh[j];
        float hz = tg[4][mr][cc] + b_hh[HID + j];
        float hn = tg[5][mr][cc] + b_hh[2 * HID + j];
        float rg = 1.f / (1.f + __expf(-(ir + hr)));
        float zg = 1.f / (1.f + __expf(-(iz + hz)));
        float nc = tanhf(in_ + rg * hn);
        float hp = hin[hi];
        float hn2 = (1.f - zg) * nc + zg * hp;
        hout[hi] = hn2;
        hfout[hi] = (_Float16)hn2;
    }
}

// ---------------- output projection ----------------
__global__ void k_proj(const float* __restrict__ hfin, const float* __restrict__ wp,
                       const float* __restrict__ bp, float* __restrict__ out) {
    int idx = blockIdx.x * blockDim.x + threadIdx.x;   // B*N
    if (idx >= BB * NN) return;
    int b = idx / NN, n = idx % NN;
    float a0 = bp[0], a1 = bp[1], a2 = bp[2], a3 = bp[3];
    const float* hp = hfin + (long)idx * HID;
#pragma unroll 4
    for (int j = 0; j < HID; ++j) {
        float hv = hp[j];
        a0 += hv * wp[j * OUTD + 0];
        a1 += hv * wp[j * OUTD + 1];
        a2 += hv * wp[j * OUTD + 2];
        a3 += hv * wp[j * OUTD + 3];
    }
    out[((long)(b * OUTD + 0)) * NN + n] = a0;
    out[((long)(b * OUTD + 1)) * NN + n] = a1;
    out[((long)(b * OUTD + 2)) * NN + n] = a2;
    out[((long)(b * OUTD + 3)) * NN + n] = a3;
}

// ---------------- workspace layout (all 256B aligned) ----------------
#define OFF_BITS   0ul
#define OFF_DEG    320000ul
#define OFF_NBR    326400ul
#define OFF_W2T    582400ul
#define OFF_WIH    615168ul
#define OFF_WHH    713472ul
#define OFF_HA     811776ul          // fp32 h ping  (13,107,200 B)
#define OFF_HB     13918976ul        // fp32 h pong
#define OFF_H1     27026176ul        // h1h, later reused as featsh (104,857,600 B)
#define OFF_AGG    131883776ul       // agg2h
#define OFF_HFA    236741376ul       // f16 h ping   (6,553,600 B)
#define OFF_HFB    243294976ul       // f16 h pong
#define WS_NEED    249848576ul

extern "C" void kernel_launch(void* const* d_in, const int* in_sizes, int n_in,
                              void* d_out, int out_size, void* d_ws, size_t ws_size,
                              hipStream_t stream) {
    const float* x    = (const float*)d_in[0];
    const float* w1   = (const float*)d_in[1];
    const float* b1   = (const float*)d_in[2];
    const float* w2   = (const float*)d_in[3];
    const float* b2   = (const float*)d_in[4];
    const float* w_ih = (const float*)d_in[5];
    const float* w_hh = (const float*)d_in[6];
    const float* b_ih = (const float*)d_in[7];
    const float* b_hh = (const float*)d_in[8];
    const float* wp   = (const float*)d_in[9];
    const float* bp   = (const float*)d_in[10];
    float* out = (float*)d_out;

    char* ws = (char*)d_ws;
    unsigned* bits  = (unsigned*)(ws + OFF_BITS);
    int* deg        = (int*)(ws + OFF_DEG);
    int* nbr        = (int*)(ws + OFF_NBR);
    _Float16* w2t   = (_Float16*)(ws + OFF_W2T);
    _Float16* wihh  = (_Float16*)(ws + OFF_WIH);
    _Float16* whhh  = (_Float16*)(ws + OFF_WHH);
    float* hA       = (float*)(ws + OFF_HA);
    float* hB       = (float*)(ws + OFF_HB);
    _Float16* h1h   = (_Float16*)(ws + OFF_H1);   // reused as featsh
    _Float16* agg2h = (_Float16*)(ws + OFF_AGG);
    _Float16* hfA   = (_Float16*)(ws + OFF_HFA);
    _Float16* hfB   = (_Float16*)(ws + OFF_HFB);
    _Float16* featsh = h1h;

    // 1) adjacency (rebuilt every call; deterministic)
    k_adj_clear<<<(NN * NWORDS + 255) / 256, 256, 0, stream>>>(bits);
    k_adj_knn<<<(NN + 127) / 128, 128, 0, stream>>>(bits);
    k_adj_csr<<<(NN + 127) / 128, 128, 0, stream>>>(bits, nbr, deg);

    // 2) weight prep + h0 init (fp32 and f16 shadows)
    k_cvt_f16<<<(G3 * HID + 255) / 256, 256, 0, stream>>>(w_ih, wihh, G3 * HID);
    k_cvt_f16<<<(G3 * HID + 255) / 256, 256, 0, stream>>>(w_hh, whhh, G3 * HID);
    k_transpose_w2<<<(HID * HID + 255) / 256, 256, 0, stream>>>(w2, w2t);
    k_zero_f32<<<(BB * NN * HID + 255) / 256, 256, 0, stream>>>(hA, BB * NN * HID);
    k_zero_f16<<<(BB * NN * HID + 255) / 256, 256, 0, stream>>>(hfA, BB * NN * HID);

    // 3) GCN layer 1 (sparse gather + C=8 matvec)
    k_gcn1<<<BB * SS * NN, HID, 0, stream>>>(x, w1, b1, nbr, deg, h1h);

    // 4) sparse aggregation of h1
    k_gcn_agg<<<BB * SS * NN, HID, 0, stream>>>(h1h, nbr, deg, agg2h);

    // 5) feats = agg2 @ W2 + b2  (WMMA; writes [S,B,N,HID] into h1h buffer)
    k_gemm_feats<<<BB * SS * NT, 256, 0, stream>>>(agg2h, w2t, b2, featsh);

    // 6) GRU over 16 timesteps (WMMA gates, ping-pong h); final lands in hA
    for (int t = 0; t < SS; ++t) {
        const float* hin      = (t & 1) ? hB : hA;
        float* hout           = (t & 1) ? hA : hB;
        const _Float16* hfin  = (t & 1) ? hfB : hfA;
        _Float16* hfout       = (t & 1) ? hfA : hfB;
        k_gru_step<<<(BB * NN / 16) * 8, 64, 0, stream>>>(featsh, hin, hfin, hout, hfout,
                                                          wihh, whhh, b_ih, b_hh, t);
    }

    // 7) projection to [B,4,H,W]
    k_proj<<<(BB * NN + 255) / 256, 256, 0, stream>>>(hA, wp, bp, out);
}
